// MultiHeadAttention_55336358642053
// MI455X (gfx1250) — matrix operand order repaired and checked by reading, not compiled, not run on hardware
//
#include <hip/hip_runtime.h>
#include <hip/hip_bf16.h>

typedef __bf16 bf16_t;
typedef __attribute__((ext_vector_type(16))) __bf16 v16bf;
typedef __attribute__((ext_vector_type(8)))  float  v8f;

union Frag {
    v16bf v;
    uint4 q[2];
};

#define WMMA_BF16(A_, B_, C_) \
    __builtin_amdgcn_wmma_f32_16x16x32_bf16(false, (A_), false, (B_), (short)0, (C_), false, false)

// ---------------------------------------------------------------------------
// CDNA5 async global->LDS staging (ASYNCcnt-tracked DMA, bypasses VGPRs).
// Generic pointers to __shared__ carry the LDS byte offset in their low 32
// bits (aperture layout), which is what the vdst LDS-address VGPR expects.
// ---------------------------------------------------------------------------
__device__ __forceinline__ uint32_t lds_off(const void* p) {
    return (uint32_t)(uintptr_t)p;
}
__device__ __forceinline__ void async_b128(uint32_t lds_byte_addr, const void* gaddr) {
    asm volatile("global_load_async_to_lds_b128 %0, %1, off"
                 :: "v"(lds_byte_addr), "v"(gaddr)
                 : "memory");
}
__device__ __forceinline__ void wait_async0() {
    asm volatile("s_wait_asynccnt 0x0" ::: "memory");
}

// ---------------------------------------------------------------------------
// f32 -> bf16 conversion (grid-stride)
// ---------------------------------------------------------------------------
__global__ __launch_bounds__(256)
void cvt_f32_bf16(const float* __restrict__ in, bf16_t* __restrict__ out, int n) {
    int i = blockIdx.x * blockDim.x + threadIdx.x;
    const int stride = gridDim.x * blockDim.x;
    for (; i < n; i += stride) out[i] = (bf16_t)in[i];
}

// ---------------------------------------------------------------------------
// C[M,N] = A[M,K] @ W[N,K]^T + bias  (torch Linear), bf16 in, f32 acc.
// Block tile 256x128, 8 waves, wave tile 64x64 (4x4 subtiles of 16x16).
// Double-buffered LDS fed by async global->LDS DMA; 16 WMMAs per K-step
// per wave overlap the async fill of the next buffer.
// ---------------------------------------------------------------------------
__global__ __launch_bounds__(256)
void gemm_nt_bf16(const bf16_t* __restrict__ A,
                  const bf16_t* __restrict__ W,
                  const float*  __restrict__ bias,
                  bf16_t* __restrict__ Cb,
                  float*  __restrict__ Cf,
                  int M, int N, int K)
{
    __shared__ alignas(16) bf16_t As[2][256][40];   // 80B stride, 16B aligned
    __shared__ alignas(16) bf16_t Ws[2][128][40];

    const int tid   = threadIdx.x;
    const int lane  = tid & 31;
    const int wave  = tid >> 5;
    const int wm    = wave >> 1;       // 0..3 (64-row strip)
    const int wn    = wave & 1;        // 0..1 (64-col strip)
    const int mlane = lane & 15;
    const int hi    = lane >> 4;       // K-half for A/B frags, M-half for C

    const int bm = blockIdx.x * 256;
    const int bn = blockIdx.y * 128;

    v8f vzero = {};
    v8f acc[4][4];
#pragma unroll
    for (int mi = 0; mi < 4; ++mi)
#pragma unroll
        for (int ni = 0; ni < 4; ++ni)
            acc[mi][ni] = vzero;

    // loader geometry: A tile 256x32 (1 row/thread), W tile 128x32 (half row)
    const int arow = tid;
    const int wrow = tid >> 1;
    const int wcol = (tid & 1) * 16;

    auto issue_tile = [&](int buf, int k0) {
        const bf16_t* ag = A + (size_t)(bm + arow) * K + k0;
        const uint32_t  al = lds_off(&As[buf][arow][0]);
        async_b128(al,      ag);
        async_b128(al + 16, ag + 8);
        async_b128(al + 32, ag + 16);
        async_b128(al + 48, ag + 24);
        const bf16_t* wg = W + (size_t)(bn + wrow) * K + k0 + wcol;
        const uint32_t  wl = lds_off(&Ws[buf][wrow][wcol]);
        async_b128(wl,      wg);
        async_b128(wl + 16, wg + 8);
    };

    issue_tile(0, 0);

    const int KT = K >> 5;
    for (int kt = 0; kt < KT; ++kt) {
        wait_async0();          // buffer kt&1 resident in LDS
        __syncthreads();
        if (kt + 1 < KT)
            issue_tile((kt + 1) & 1, (kt + 1) << 5);   // overlaps the WMMAs below

        const int buf = kt & 1;
        Frag af[4], wf[4];
#pragma unroll
        for (int mi = 0; mi < 4; ++mi) {
            const int r = wm * 64 + mi * 16 + mlane;
            // 16-bit A 16x32 layout: lane half -> K base 0/8, elems 8..15 at K+16
            af[mi].q[0] = *(const uint4*)&As[buf][r][hi * 8];
            af[mi].q[1] = *(const uint4*)&As[buf][r][16 + hi * 8];
        }
#pragma unroll
        for (int ni = 0; ni < 4; ++ni) {
            const int c = wn * 64 + ni * 16 + mlane;
            // 16-bit B 32x16 layout: lane half -> K base 0/8, elems 8..15 at K+16
            wf[ni].q[0] = *(const uint4*)&Ws[buf][c][hi * 8];
            wf[ni].q[1] = *(const uint4*)&Ws[buf][c][16 + hi * 8];
        }
#pragma unroll
        for (int mi = 0; mi < 4; ++mi)
#pragma unroll
            for (int ni = 0; ni < 4; ++ni)
                acc[mi][ni] = WMMA_BF16(af[mi].v, wf[ni].v, acc[mi][ni]);
        __syncthreads();
    }

#pragma unroll
    for (int mi = 0; mi < 4; ++mi) {
#pragma unroll
        for (int ni = 0; ni < 4; ++ni) {
            const int col = bn + wn * 64 + ni * 16 + mlane;
            const float bv = bias ? bias[col] : 0.0f;
#pragma unroll
            for (int r = 0; r < 8; ++r) {
                const int row = bm + wm * 64 + mi * 16 + hi * 8 + r;
                const float v = acc[mi][ni][r] + bv;
                const size_t idx = (size_t)row * N + col;
                if (Cf) Cf[idx] = v;
                if (Cb) Cb[idx] = (bf16_t)v;
            }
        }
    }
}

// ---------------------------------------------------------------------------
// Flash attention per (batch, head, 128-row q tile). 8 waves, 16 q-rows/wave.
// kv tiles of 64, online softmax, causal additive -10000 mask (== reference).
// Q/K tiles staged via async DMA; V transposed manually into [d][kv].
// Writes ctx directly in merged-head [B,S,D] layout.
// ---------------------------------------------------------------------------
__global__ __launch_bounds__(256)
void flash_attn_bf16(const bf16_t* __restrict__ Qp,
                     const bf16_t* __restrict__ Kp,
                     const bf16_t* __restrict__ Vp,
                     bf16_t* __restrict__ Ctx)
{
    constexpr int S = 2048, D = 1024, DEP = 64;

    __shared__ alignas(16) bf16_t Qs[128][72];
    __shared__ alignas(16) bf16_t Ks[64][72];
    __shared__ alignas(16) bf16_t Vt[64][72];     // transposed: [d][kv]
    __shared__ alignas(16) bf16_t Ps[8][16][72];  // per-wave P tile (C->A relayout)

    const int tid   = threadIdx.x;
    const int lane  = tid & 31;
    const int wv    = tid >> 5;
    const int mlane = lane & 15;
    const int hi    = lane >> 4;

    const int q0   = blockIdx.x * 128;
    const int h    = blockIdx.y;
    const int b    = blockIdx.z;
    const size_t bs   = (size_t)b * S;
    const int    hoff = h * DEP;

    // stage Q tile [128 x 64] via async DMA
    {
        const int row = tid >> 1;
        const int col = (tid & 1) * 32;
        const bf16_t* g = Qp + (bs + q0 + row) * D + hoff + col;
        const uint32_t l = lds_off(&Qs[row][col]);
        async_b128(l,      g);
        async_b128(l + 16, g + 8);
        async_b128(l + 32, g + 16);
        async_b128(l + 48, g + 24);
    }
    wait_async0();
    __syncthreads();

    Frag qf[2];
#pragma unroll
    for (int c = 0; c < 2; ++c) {
        qf[c].q[0] = *(const uint4*)&Qs[wv * 16 + mlane][c * 32 + hi * 8];
        qf[c].q[1] = *(const uint4*)&Qs[wv * 16 + mlane][c * 32 + 16 + hi * 8];
    }

    v8f vzero = {};
    v8f o[4] = {vzero, vzero, vzero, vzero};
    float mrow[8], lrow[8];
#pragma unroll
    for (int r = 0; r < 8; ++r) { mrow[r] = -1e30f; lrow[r] = 0.0f; }

    const int jend = q0 + 128;   // causal bound
    for (int j = 0; j < jend; j += 64) {
        // K tile via async DMA; V tile loaded to regs and transposed into Vt
        {
            const int row = tid >> 2;          // kv 0..63
            const int col = (tid & 3) * 16;    // d
            const bf16_t* gk = Kp + (bs + j + row) * D + hoff + col;
            const uint32_t kl = lds_off(&Ks[row][col]);
            async_b128(kl,      gk);
            async_b128(kl + 16, gk + 8);

            const uint4* gv = (const uint4*)(Vp + (bs + j + row) * D + hoff + col);
            union { uint4 q[2]; bf16_t e[16]; } tmp;
            tmp.q[0] = gv[0];
            tmp.q[1] = gv[1];
#pragma unroll
            for (int e = 0; e < 16; ++e)
                Vt[col + e][row] = tmp.e[e];
        }
        wait_async0();
        __syncthreads();

        // scores: S = (Q K^T) * 1/sqrt(64)  (contraction 64 = 2 chained WMMAs)
        v8f s[4];
#pragma unroll
        for (int ni = 0; ni < 4; ++ni) {
            const int n = ni * 16 + mlane;
            Frag kb0, kb1;
            kb0.q[0] = *(const uint4*)&Ks[n][hi * 8];
            kb0.q[1] = *(const uint4*)&Ks[n][16 + hi * 8];
            kb1.q[0] = *(const uint4*)&Ks[n][32 + hi * 8];
            kb1.q[1] = *(const uint4*)&Ks[n][32 + 16 + hi * 8];
            v8f sc = WMMA_BF16(qf[0].v, kb0.v, vzero);
            sc = WMMA_BF16(qf[1].v, kb1.v, sc);
            s[ni] = sc;
        }

        // mask + scale + running row max (16-lane halves via shfl_xor)
        float mnew[8];
#pragma unroll
        for (int r = 0; r < 8; ++r) {
            const int qrow = q0 + wv * 16 + hi * 8 + r;
            float mx = mrow[r];
#pragma unroll
            for (int ni = 0; ni < 4; ++ni) {
                const int kcol = j + ni * 16 + mlane;
                float val = s[ni][r] * 0.125f + ((kcol > qrow) ? -10000.0f : 0.0f);
                s[ni][r] = val;
                mx = fmaxf(mx, val);
            }
#pragma unroll
            for (int off = 8; off >= 1; off >>= 1)
                mx = fmaxf(mx, __shfl_xor(mx, off, 32));
            mnew[r] = mx;
        }

        // exponentiate, row sums, rescale accumulators
#pragma unroll
        for (int r = 0; r < 8; ++r) {
            const float corr = __expf(mrow[r] - mnew[r]);
            mrow[r] = mnew[r];
            float rs = 0.0f;
#pragma unroll
            for (int ni = 0; ni < 4; ++ni) {
                const float p = __expf(s[ni][r] - mnew[r]);
                s[ni][r] = p;
                rs += p;
            }
#pragma unroll
            for (int off = 8; off >= 1; off >>= 1)
                rs += __shfl_xor(rs, off, 32);
            lrow[r] = lrow[r] * corr + rs;
#pragma unroll
            for (int nd = 0; nd < 4; ++nd)
                o[nd][r] = o[nd][r] * corr;
        }

        // P: C-layout -> LDS -> A-layout fragments (wave-private region)
#pragma unroll
        for (int ni = 0; ni < 4; ++ni)
#pragma unroll
            for (int r = 0; r < 8; ++r)
                Ps[wv][hi * 8 + r][ni * 16 + mlane] = (bf16_t)s[ni][r];

        Frag pf[2];
#pragma unroll
        for (int c = 0; c < 2; ++c) {
            pf[c].q[0] = *(const uint4*)&Ps[wv][mlane][c * 32 + hi * 8];
            pf[c].q[1] = *(const uint4*)&Ps[wv][mlane][c * 32 + 16 + hi * 8];
        }

        // O += P @ V   (contraction kv=64 -> 2 chained WMMAs per 16-col d tile)
#pragma unroll
        for (int nd = 0; nd < 4; ++nd) {
            const int dcol = nd * 16 + mlane;
            Frag vf0, vf1;
            vf0.q[0] = *(const uint4*)&Vt[dcol][hi * 8];
            vf0.q[1] = *(const uint4*)&Vt[dcol][16 + hi * 8];
            vf1.q[0] = *(const uint4*)&Vt[dcol][32 + hi * 8];
            vf1.q[1] = *(const uint4*)&Vt[dcol][32 + 16 + hi * 8];
            o[nd] = WMMA_BF16(pf[0].v, vf0.v, o[nd]);
            o[nd] = WMMA_BF16(pf[1].v, vf1.v, o[nd]);
        }
        __syncthreads();
    }

    // normalize and store merged-head context
#pragma unroll
    for (int nd = 0; nd < 4; ++nd) {
#pragma unroll
        for (int r = 0; r < 8; ++r) {
            const int qrow = q0 + wv * 16 + hi * 8 + r;
            const float val = o[nd][r] / lrow[r];
            Ctx[(bs + qrow) * D + hoff + nd * 16 + mlane] = (bf16_t)val;
        }
    }
}

// ---------------------------------------------------------------------------
// launcher
// ---------------------------------------------------------------------------
extern "C" void kernel_launch(void* const* d_in, const int* in_sizes, int n_in,
                              void* d_out, int out_size, void* d_ws, size_t ws_size,
                              hipStream_t stream)
{
    constexpr int B = 4, S = 2048, D = 1024, H = 16;
    (void)in_sizes; (void)n_in; (void)out_size; (void)ws_size;

    const float* q    = (const float*)d_in[0];
    const float* k    = (const float*)d_in[1];
    const float* v    = (const float*)d_in[2];
    // d_in[3] = mask: reproduced analytically (triu k=1 -> kcol > qrow)
    const float* Wq_w = (const float*)d_in[4];
    const float* Wq_b = (const float*)d_in[5];
    const float* Wk_w = (const float*)d_in[6];
    const float* Wk_b = (const float*)d_in[7];
    const float* Wv_w = (const float*)d_in[8];
    const float* Wv_b = (const float*)d_in[9];
    const float* Wd_w = (const float*)d_in[10];
    const float* Wd_b = (const float*)d_in[11];

    const size_t eBSD = (size_t)B * S * D;   // 8,388,608
    const size_t eDD  = (size_t)D * D;       // 1,048,576

    bf16_t* ws  = (bf16_t*)d_ws;
    bf16_t* qb  = ws;  ws += eBSD;
    bf16_t* kb  = ws;  ws += eBSD;
    bf16_t* vb  = ws;  ws += eBSD;
    bf16_t* wqb = ws;  ws += eDD;
    bf16_t* wkb = ws;  ws += eDD;
    bf16_t* wvb = ws;  ws += eDD;
    bf16_t* wdb = ws;  ws += eDD;
    bf16_t* Qp  = ws;  ws += eBSD;
    bf16_t* Kp  = ws;  ws += eBSD;
    bf16_t* Vp  = ws;  ws += eBSD;
    bf16_t* Ctx = ws;  ws += eBSD;

    // f32 -> bf16 staging
    cvt_f32_bf16<<<4096, 256, 0, stream>>>(q,    qb,  (int)eBSD);
    cvt_f32_bf16<<<4096, 256, 0, stream>>>(k,    kb,  (int)eBSD);
    cvt_f32_bf16<<<4096, 256, 0, stream>>>(v,    vb,  (int)eBSD);
    cvt_f32_bf16<<<1024, 256, 0, stream>>>(Wq_w, wqb, (int)eDD);
    cvt_f32_bf16<<<1024, 256, 0, stream>>>(Wk_w, wkb, (int)eDD);
    cvt_f32_bf16<<<1024, 256, 0, stream>>>(Wv_w, wvb, (int)eDD);
    cvt_f32_bf16<<<1024, 256, 0, stream>>>(Wd_w, wdb, (int)eDD);

    // QKV projections: [8192,1024] @ [1024,1024]^T (block tile 256x128)
    const dim3 ggrid(B * S / 256, D / 128);
    gemm_nt_bf16<<<ggrid, 256, 0, stream>>>(qb, wqb, Wq_b, Qp, nullptr, B * S, D, D);
    gemm_nt_bf16<<<ggrid, 256, 0, stream>>>(kb, wkb, Wk_b, Kp, nullptr, B * S, D, D);
    gemm_nt_bf16<<<ggrid, 256, 0, stream>>>(vb, wvb, Wv_b, Vp, nullptr, B * S, D, D);

    // fused causal attention, ctx in merged-head layout
    flash_attn_bf16<<<dim3(S / 128, H, B), 256, 0, stream>>>(Qp, Kp, Vp, Ctx);

    // output projection straight to f32 d_out
    gemm_nt_bf16<<<ggrid, 256, 0, stream>>>(Ctx, wdb, Wd_b, nullptr, (float*)d_out, B * S, D, D);
}
